// Pooler_6837587936138
// MI455X (gfx1250) — matrix operand order repaired and checked by reading, not compiled, not run on hardware
//
#include <hip/hip_runtime.h>

// Problem constants (match reference setup_inputs)
#define BB 8
#define TT 2048
#define DD 1024
#define HH 8
#define SS 128

typedef __attribute__((ext_vector_type(16))) __bf16 v16bf;
typedef __attribute__((ext_vector_type(8)))  float  v8f;

// ---------------------------------------------------------------------------
// CDNA5 async global->LDS copies (inline asm per CDNA5 ISA;
// the clang builtin arity differs across toolchains, asm is portable).
// VDST = VGPR holding LDS byte address, VADDR = 64-bit global address.
// ---------------------------------------------------------------------------
__device__ __forceinline__ void async_g2l_b128(void* lds, const void* g) {
    asm volatile("global_load_async_to_lds_b128 %0, %1, off"
                 :: "v"((unsigned)(uintptr_t)lds), "v"(g) : "memory");
}
__device__ __forceinline__ void async_g2l_b32(void* lds, const void* g) {
    asm volatile("global_load_async_to_lds_b32 %0, %1, off"
                 :: "v"((unsigned)(uintptr_t)lds), "v"(g) : "memory");
}
__device__ __forceinline__ void wait_async0() {
    asm volatile("s_wait_asynccnt 0x0" ::: "memory");
}

// ---------------------------------------------------------------------------
// Kernel A: logits[(b*T+t)*H + h] = features[b,t,:]·key_w[h,:] + key_b[h]
// key_w is staged once per block: async-copied fp32 to LDS, converted to a
// 16-row bf16 LDS image (rows 8..15 zero).  Each wave computes a 16x16 C tile
// (16 rows of B*T x 16 heads) with v_wmma_f32_16x16x32_bf16; B-fragments are
// unconditional ds_load_b128 pairs, A-fragments stream from HBM.
// ---------------------------------------------------------------------------
__global__ __launch_bounds__(256)
void pooler_logits_wmma(const float* __restrict__ features, // [B*T, D]
                        const float* __restrict__ key_w,    // [H, D]
                        const float* __restrict__ key_b,    // [H]
                        float* __restrict__ logits)         // [B*T, H]
{
    __shared__ __align__(16) float  kw_f32[HH * DD];    // 32 KB staging
    __shared__ __align__(32) __bf16 kw_bf16[16 * DD];   // 32 KB, rows 8..15 = 0

    const int tid  = threadIdx.x;
    const int lane = tid & 31;
    const int wave = tid >> 5;
    const int tile = blockIdx.x * 8 + wave;            // 16-row tile of B*T
    const long rowBase = (long)tile * 16;

    // ---- stage key_w: async global->LDS (8192 floats = 2048 b128 chunks) ----
    for (int i = tid; i < (HH * DD) / 4; i += 256)
        async_g2l_b128(&kw_f32[i * 4], &key_w[i * 4]);
    wait_async0();
    __syncthreads();
    // fp32 -> bf16 once per block; zero-pad heads 8..15 so every lane of the
    // B-fragment load is unconditional (no per-iteration EXEC masking).
    for (int i = tid; i < HH * DD; i += 256)
        kw_bf16[i] = (__bf16)kw_f32[i];
    for (int i = tid; i < HH * DD; i += 256)
        kw_bf16[HH * DD + i] = (__bf16)0.0f;
    __syncthreads();

    const int half = lane >> 4;                        // lane half (0/1)
    const int l15  = lane & 15;

    v8f c = {};                                        // 16x16 f32 accumulator

    const float*  Arow = features + (rowBase + l15) * (long)DD;
    const __bf16* Brow = &kw_bf16[l15 * DD];           // B = key_w^T (padded)

#pragma unroll 2
    for (int k0 = 0; k0 < DD; k0 += 32) {
        // A fragment (contiguous k order, same as B):
        //   lanes 0-15 : K = k0..k0+15; lanes 16-31: K = k0+16..k0+31
        const float* a0 = Arow + k0 + half * 16;
        const float* a1 = a0 + 8;
        v16bf a;
#pragma unroll
        for (int i = 0; i < 8; ++i) {
            a[i]     = (__bf16)a0[i];
            a[i + 8] = (__bf16)a1[i];
        }

        // B fragment (contiguous k order, same as A) straight from LDS bf16 image:
        //   lanes 0-15: N=lane, K=k0..k0+15 ; lanes 16-31: N=lane-16, K=+16
        v16bf b = *(const v16bf*)(Brow + k0 + half * 16);

        if (k0 + 32 < DD) __builtin_prefetch(a0 + 32, 0, 3);  // near-cache prefetch

        c = __builtin_amdgcn_wmma_f32_16x16x32_bf16(
                /*neg_a=*/false, a, /*neg_b=*/false, b,
                /*c_mod=*/(short)0, c, /*reuse_a=*/false, /*reuse_b=*/false);
    }

    // C/D layout: VGPR r -> M = r + 8*half, N = lane&15.  Keep heads 0..7.
    if (l15 < HH) {
        const float bias = key_b[l15];
#pragma unroll
        for (int r = 0; r < 8; ++r) {
            const int m = r + half * 8;
            logits[(rowBase + m) * HH + l15] = c[r] + bias;
        }
    }
}

// ---------------------------------------------------------------------------
// Kernel B: per-(b,s) masked softmax over the span (width <= 32) and the
// weighted sum of per-head feature slices.  exp(-1e5 - max) underflows to 0
// in fp32, so span-only softmax matches the reference exactly.
// ---------------------------------------------------------------------------
__global__ __launch_bounds__(256)
void pooler_span_pool(const float* __restrict__ features,   // [B*T, D]
                      const long long* __restrict__ begins, // [B, S]
                      const long long* __restrict__ ends,   // [B, S]
                      const float* __restrict__ logits,     // [B*T, H]
                      float* __restrict__ out)              // [B*S, D]
{
    __shared__ float w[32][HH];

    const int bs  = blockIdx.x;          // b*S + s
    const int b   = bs / SS;
    const int tid = threadIdx.x;

    const int begin = (int)begins[bs];
    const int end   = (int)ends[bs];
    const int width = end - begin;       // 1..32 guaranteed by setup

    // Stage span logits into LDS via async copy (width*H <= 256, one b32/lane)
    if (tid < width * HH) {
        const int t0 = tid >> 3;
        const int h  = tid & 7;
        async_g2l_b32(&w[t0][h],
                      &logits[((long)b * TT + begin + t0) * HH + h]);
    }
    wait_async0();
    __syncthreads();

    // Per-head softmax over the span (threads 0..7, <=96 scalar ops each)
    if (tid < HH) {
        float m = -3.4e38f;
        for (int t0 = 0; t0 < width; ++t0) m = fmaxf(m, w[t0][tid]);
        float s = 0.0f;
        for (int t0 = 0; t0 < width; ++t0) s += __expf(w[t0][tid] - m);
        const float inv = 1.0f / s;
        for (int t0 = 0; t0 < width; ++t0)
            w[t0][tid] = __expf(w[t0][tid] - m) * inv;
    }
    __syncthreads();

    // out[bs, j] = sum_t w[t, j/128] * features[b, begin+t, j], j = 0..1023
    const float* src = features + ((long)b * TT + begin) * DD;
#pragma unroll
    for (int rep = 0; rep < 4; ++rep) {
        const int j = tid + rep * 256;
        const int h = j >> 7;
        float acc = 0.0f;
        for (int t0 = 0; t0 < width; ++t0)
            acc += w[t0][h] * src[(long)t0 * DD + j];
        out[(long)bs * DD + j] = acc;
    }
}

// ---------------------------------------------------------------------------
extern "C" void kernel_launch(void* const* d_in, const int* in_sizes, int n_in,
                              void* d_out, int out_size, void* d_ws, size_t ws_size,
                              hipStream_t stream)
{
    const float*     features = (const float*)d_in[0];      // [B,T,D] fp32
    const long long* begins   = (const long long*)d_in[1];  // [B,S] int64
    const long long* ends     = (const long long*)d_in[2];  // [B,S] int64
    const float*     key_w    = (const float*)d_in[3];      // [H,D] fp32
    const float*     key_b    = (const float*)d_in[4];      // [H]   fp32
    float*           out      = (float*)d_out;              // [B,S,D] fp32
    float*           logits   = (float*)d_ws;               // 512 KB scratch

    // 1024 row-tiles of 16, 8 waves (one tile each) per 256-thread block
    const int tiles   = (BB * TT) / 16;      // 1024
    const int blocksA = tiles / 8;           // 128
    pooler_logits_wmma<<<blocksA, 256, 0, stream>>>(features, key_w, key_b, logits);

    // one block per (b, s) span
    pooler_span_pool<<<BB * SS, 256, 0, stream>>>(features, begins, ends, logits, out);
}
